// NodeEncoder_72971494359603
// MI455X (gfx1250) — matrix operand order repaired and checked by reading, not compiled, not run on hardware
//
#include <hip/hip_runtime.h>
#include <hip/hip_bf16.h>

// ---------------------------------------------------------------------------
// GIN forward (5 layers) for gfx1250: gather/scatter in fp32 with atomics,
// GEMMs via v_wmma_f32_16x16x32_bf16 (bf16 inputs, f32 accumulate).
// A-tiles double-buffered in LDS via global_load_async_to_lds_b128, pipelined
// against the WMMA pipe (issue next tile -> compute current -> wait ASYNCcnt).
// ---------------------------------------------------------------------------

#define D_DIM 256
#define L_LAYERS 5
#define BN_EPS 1e-5f

typedef __attribute__((ext_vector_type(16))) __bf16 v16bf;
typedef __attribute__((ext_vector_type(8)))  float  v8f;

static __host__ __device__ inline int cdiv(int a, int b) { return (a + b - 1) / b; }

// --------------------------- elementwise kernels ---------------------------

__global__ void k_cvt_f32_to_bf16(const float* __restrict__ in,
                                  __hip_bfloat16* __restrict__ out, int n) {
    int i = blockIdx.x * blockDim.x + threadIdx.x;
    int stride = gridDim.x * blockDim.x;
    for (; i < n; i += stride) out[i] = __float2bfloat16(in[i]);
}

// out[i,c] = x_emb1[x[i,0],c] + x_emb2[x[i,1],c]
__global__ void k_init_out(const int* __restrict__ x,
                           const float* __restrict__ xe1,
                           const float* __restrict__ xe2,
                           float* __restrict__ out, int n) {
    int i = blockIdx.x;
    if (i >= n) return;
    int c = threadIdx.x;
    int a = x[2 * i], b = x[2 * i + 1];
    out[i * D_DIM + c] = xe1[a * D_DIM + c] + xe2[b * D_DIM + c];
}

// agg[i,c] = out[i,c] + ee1[l,4,c] + ee2[l,0,c]    (self-loop edge)
__global__ void k_agg_init(const float* __restrict__ out,
                           const float* __restrict__ ee1,
                           const float* __restrict__ ee2,
                           float* __restrict__ agg, int n, int l) {
    int i = blockIdx.x;
    if (i >= n) return;
    int c = threadIdx.x;
    float es = ee1[(l * 6 + 4) * D_DIM + c] + ee2[(l * 3 + 0) * D_DIM + c];
    agg[i * D_DIM + c] = out[i * D_DIM + c] + es;
}

// agg[dst,c] += out[src,c] + ee1[l,a0,c] + ee2[l,a1,c]
__global__ void k_scatter(const int* __restrict__ ei,   // [2,E]
                          const int* __restrict__ ea,   // [E,2]
                          const float* __restrict__ out,
                          const float* __restrict__ ee1,
                          const float* __restrict__ ee2,
                          float* __restrict__ agg, int e_cnt, int l) {
    int e = blockIdx.x;
    if (e >= e_cnt) return;
    int c = threadIdx.x;
    int src = ei[e], dst = ei[e_cnt + e];
    int a0 = ea[2 * e], a1 = ea[2 * e + 1];
    float msg = out[src * D_DIM + c] + ee1[(l * 6 + a0) * D_DIM + c]
              + ee2[(l * 3 + a1) * D_DIM + c];
    atomicAdd(&agg[dst * D_DIM + c], msg);
}

// --------------------------------- GEMM ------------------------------------
// C[M,NC] = A[M,K] (bf16) * B[K,NC] (bf16) + bias, optional relu->bf16 out.
// Block: 256 threads (8 waves). Tile: BM=128 x BN=64, BK=32, double-buffered.
// Each wave computes a 32x32 patch = 2x2 v_wmma_f32_16x16x32_bf16 tiles.

#define BM 128
#define BN 64
#define BK 32

union FragBF { uint4 u[2]; v16bf v; };
union Pack8  { uint4 u; unsigned short h[8]; };

// LDS byte-offset of a generic pointer into __shared__ (low 32 bits of the
// LDS-aperture generic address are the LDS byte address).
static __device__ inline unsigned lds_off(const void* p) {
    return (unsigned)(unsigned long long)(size_t)p;
}

template <bool RELU_BF16_OUT>
__global__ __launch_bounds__(256) void k_gemm_wmma(
    const __hip_bfloat16* __restrict__ A,   // [M,K]
    const __hip_bfloat16* __restrict__ B,   // [K,NC]
    const float* __restrict__ bias,         // [NC]
    float* __restrict__ outF,               // used when !RELU_BF16_OUT
    __hip_bfloat16* __restrict__ outB,      // used when RELU_BF16_OUT
    int M, int K, int NC) {

    __shared__ __align__(16) unsigned short sA[2][BM][BK];   // A tiles (row-major)
    __shared__ __align__(16) unsigned short sBT[2][BN][BK];  // B tiles (transposed)

    const int tid  = threadIdx.x;
    const int row0 = blockIdx.x * BM;
    const int col0 = blockIdx.y * BN;

    const int wave = tid >> 5;       // 0..7
    const int lane = tid & 31;
    const int wm   = wave & 3;       // 4 wave-rows  (32 rows each)
    const int wn   = wave >> 2;      // 2 wave-cols  (32 cols each)
    const int half = lane >> 4;      // lane half
    const int lm   = lane & 15;

    const bool fullTile = (row0 + BM) <= M;

    // Per-thread A-staging geometry: exactly 2 x 16B chunks per thread.
    const int ar0 = tid >> 2;        // tile row for chunk 0 (0..63)
    const int ar1 = 64 + (tid >> 2); // tile row for chunk 1 (64..127)
    const int ap  = (tid & 3) * 8;   // element offset of 16B chunk within row

    // --- stage A tile kt into LDS buffer buf ---
    auto stageA = [&](int kt, int buf) {
        if (fullTile) {
            unsigned d0 = lds_off(&sA[buf][ar0][ap]);
            unsigned d1 = lds_off(&sA[buf][ar1][ap]);
            unsigned long long s0 = (unsigned long long)(size_t)(
                A + (size_t)(row0 + ar0) * K + kt * BK + ap);
            unsigned long long s1 = (unsigned long long)(size_t)(
                A + (size_t)(row0 + ar1) * K + kt * BK + ap);
            asm volatile("global_load_async_to_lds_b128 %0, %1, off"
                         :: "v"(d0), "v"(s0) : "memory");
            asm volatile("global_load_async_to_lds_b128 %0, %1, off"
                         :: "v"(d1), "v"(s1) : "memory");
        } else {
            int rows[2] = {ar0, ar1};
#pragma unroll
            for (int i = 0; i < 2; ++i) {
                int grow = row0 + rows[i];
                uint4 val = make_uint4(0, 0, 0, 0);
                if (grow < M) {
                    val = *reinterpret_cast<const uint4*>(
                        A + (size_t)grow * K + kt * BK + ap);
                }
                *reinterpret_cast<uint4*>(&sA[buf][rows[i]][ap]) = val;
            }
        }
    };

    // --- stage B tile kt (transposed) into LDS buffer buf ---
    const int bk = tid >> 3;          // 0..31
    const int bj = (tid & 7) * 8;     // col group
    auto stageB = [&](int kt, int buf) {
        Pack8 pk;
        pk.u = *reinterpret_cast<const uint4*>(
            B + (size_t)(kt * BK + bk) * NC + col0 + bj);
#pragma unroll
        for (int j = 0; j < 8; ++j) sBT[buf][bj + j][bk] = pk.h[j];
    };

    v8f acc[2][2] = {};
    const int ksteps = K / BK;

    // prologue: fill buffer 0
    stageA(0, 0);
    stageB(0, 0);
    if (fullTile) asm volatile("s_wait_asynccnt 0x0" ::: "memory");
    __syncthreads();

    for (int kt = 0; kt < ksteps; ++kt) {
        const int cur = kt & 1;
        const int nxt = cur ^ 1;

        // issue next tile's copies before consuming the current one
        if (kt + 1 < ksteps) {
            stageA(kt + 1, nxt);
            stageB(kt + 1, nxt);
        }

        // ---- fragments from current buffer ----
        FragBF fa[2], fb[2];
        const int h8 = half * 8;
#pragma unroll
        for (int s = 0; s < 2; ++s) {
            int r = wm * 32 + s * 16 + lm;
            fa[s].u[0] = *reinterpret_cast<const uint4*>(&sA[cur][r][h8]);
            fa[s].u[1] = *reinterpret_cast<const uint4*>(&sA[cur][r][16 + h8]);
        }
        const int koff = half * 8;
#pragma unroll
        for (int t = 0; t < 2; ++t) {
            int cL = wn * 32 + t * 16 + lm;
            fb[t].u[0] = *reinterpret_cast<const uint4*>(&sBT[cur][cL][koff]);
            fb[t].u[1] = *reinterpret_cast<const uint4*>(&sBT[cur][cL][koff + 16]);
        }
#pragma unroll
        for (int s = 0; s < 2; ++s)
#pragma unroll
            for (int t = 0; t < 2; ++t)
                acc[s][t] = __builtin_amdgcn_wmma_f32_16x16x32_bf16(
                    false, fa[s].v, false, fb[t].v, (short)0, acc[s][t],
                    false, false);

        // drain this iteration's async copies, then sync for next iteration
        if (fullTile && kt + 1 < ksteps)
            asm volatile("s_wait_asynccnt 0x0" ::: "memory");
        __syncthreads();
    }

    // ---- epilogue: C layout => lane half selects M+8, reg r = row ----
#pragma unroll
    for (int s = 0; s < 2; ++s) {
#pragma unroll
        for (int t = 0; t < 2; ++t) {
            int gc = col0 + wn * 32 + t * 16 + lm;
            float bv = bias[gc];
#pragma unroll
            for (int r = 0; r < 8; ++r) {
                int gr = row0 + wm * 32 + s * 16 + half * 8 + r;
                if (gr < M) {
                    float v = acc[s][t][r] + bv;
                    if (RELU_BF16_OUT) {
                        v = v > 0.f ? v : 0.f;
                        outB[(size_t)gr * NC + gc] = __float2bfloat16(v);
                    } else {
                        outF[(size_t)gr * NC + gc] = v;
                    }
                }
            }
        }
    }
}

// ------------------------------- BatchNorm ---------------------------------

__global__ void k_zero(float* p, int n) {
    int i = blockIdx.x * blockDim.x + threadIdx.x;
    if (i < n) p[i] = 0.f;
}

#define BN_ROWS 256
__global__ void k_bn_reduce(const float* __restrict__ h,
                            float* __restrict__ sums, int n) {
    int c  = threadIdx.x;                 // column 0..255
    int r0 = blockIdx.x * BN_ROWS;
    float s = 0.f, s2 = 0.f;
    int rend = min(r0 + BN_ROWS, n);
    for (int r = r0; r < rend; ++r) {
        float v = h[(size_t)r * D_DIM + c];
        s += v; s2 += v * v;
    }
    atomicAdd(&sums[c], s);
    atomicAdd(&sums[D_DIM + c], s2);
}

__global__ void k_bn_apply(const float* __restrict__ h,
                           const float* __restrict__ sums,
                           const float* __restrict__ gamma,
                           const float* __restrict__ beta,
                           float* __restrict__ out, int n, int l) {
    int c  = threadIdx.x;
    int r0 = blockIdx.x * BN_ROWS;
    float inv_n = 1.0f / (float)n;
    float mean = sums[c] * inv_n;
    float var  = sums[D_DIM + c] * inv_n - mean * mean;
    float sc   = gamma[l * D_DIM + c] * rsqrtf(var + BN_EPS);
    float sh   = beta[l * D_DIM + c] - mean * sc;
    int rend = min(r0 + BN_ROWS, n);
    for (int r = r0; r < rend; ++r) {
        float v = h[(size_t)r * D_DIM + c] * sc + sh;
        out[(size_t)r * D_DIM + c] = v > 0.f ? v : 0.f;
    }
}

// -------------------------------- launcher ---------------------------------

extern "C" void kernel_launch(void* const* d_in, const int* in_sizes, int n_in,
                              void* d_out, int out_size, void* d_ws, size_t ws_size,
                              hipStream_t stream) {
    const int*   x     = (const int*)  d_in[0];   // [N,2]
    const int*   ei    = (const int*)  d_in[1];   // [2,E]
    const int*   ea    = (const int*)  d_in[2];   // [E,2]
    const float* xe1   = (const float*)d_in[3];   // [120,D]
    const float* xe2   = (const float*)d_in[4];   // [3,D]
    const float* ee1   = (const float*)d_in[5];   // [L,6,D]
    const float* ee2   = (const float*)d_in[6];   // [L,3,D]
    const float* W1    = (const float*)d_in[7];   // [L,D,2D]
    const float* b1    = (const float*)d_in[8];   // [L,2D]
    const float* W2    = (const float*)d_in[9];   // [L,2D,D]
    const float* b2    = (const float*)d_in[10];  // [L,D]
    const float* gamma = (const float*)d_in[11];  // [L-1,D]
    const float* beta  = (const float*)d_in[12];  // [L-1,D]

    const int N = in_sizes[0] / 2;
    const int E = in_sizes[1] / 2;
    const int K1 = D_DIM, N1 = 2 * D_DIM;          // GEMM1: [N,256]x[256,512]
    const int K2 = 2 * D_DIM, N2 = D_DIM;          // GEMM2: [N,512]x[512,256]

    // ---- workspace carve-up (bytes) ----
    char* ws = (char*)d_ws;
    size_t szND4  = (size_t)N * D_DIM * 4;         // 102.4 MB
    size_t szND2  = (size_t)N * D_DIM * 2;         // 51.2 MB
    size_t szN2D2 = (size_t)N * 2 * D_DIM * 2;     // 102.4 MB
    size_t szW    = (size_t)L_LAYERS * D_DIM * 2 * D_DIM * 2;  // 1.31 MB

    float*          outBuf = (float*)(ws);                     // running node feats
    float*          aggF   = (float*)(ws + szND4);             // agg fp32; reused as h fp32
    __hip_bfloat16* aggBf  = (__hip_bfloat16*)(ws + 2 * szND4);
    __hip_bfloat16* h1Bf   = (__hip_bfloat16*)(ws + 2 * szND4 + szND2);
    __hip_bfloat16* W1bf   = (__hip_bfloat16*)(ws + 2 * szND4 + szND2 + szN2D2);
    __hip_bfloat16* W2bf   = (__hip_bfloat16*)(ws + 2 * szND4 + szND2 + szN2D2 + szW);
    float*          sums   = (float*)(ws + 2 * szND4 + szND2 + szN2D2 + 2 * szW);

    // ---- one-time weight conversion + node embedding ----
    int nW = L_LAYERS * D_DIM * 2 * D_DIM;
    k_cvt_f32_to_bf16<<<1024, 256, 0, stream>>>(W1, W1bf, nW);
    k_cvt_f32_to_bf16<<<1024, 256, 0, stream>>>(W2, W2bf, nW);
    k_init_out<<<N, D_DIM, 0, stream>>>(x, xe1, xe2, outBuf, N);

    dim3 g1(cdiv(N, BM), N1 / BN);   // 782 x 8
    dim3 g2(cdiv(N, BM), N2 / BN);   // 782 x 4

    for (int l = 0; l < L_LAYERS; ++l) {
        // message passing: self-loop init + edge scatter-add
        k_agg_init<<<N, D_DIM, 0, stream>>>(outBuf, ee1, ee2, aggF, N, l);
        k_scatter<<<E, D_DIM, 0, stream>>>(ei, ea, outBuf, ee1, ee2, aggF, E, l);
        // agg -> bf16
        k_cvt_f32_to_bf16<<<2048, 256, 0, stream>>>(aggF, aggBf, N * D_DIM);
        // h1 = relu(agg @ W1[l] + b1[l])  -> bf16
        k_gemm_wmma<true><<<g1, 256, 0, stream>>>(
            aggBf, W1bf + (size_t)l * D_DIM * N1, b1 + (size_t)l * N1,
            nullptr, h1Bf, N, K1, N1);
        // h = h1 @ W2[l] + b2[l]  -> fp32 (to aggF, or d_out for last layer)
        float* hDst = (l == L_LAYERS - 1) ? (float*)d_out : aggF;
        k_gemm_wmma<false><<<g2, 256, 0, stream>>>(
            h1Bf, W2bf + (size_t)l * 2 * D_DIM * N2, b2 + (size_t)l * N2,
            hDst, nullptr, N, K2, N2);
        // batchnorm + relu for all but last layer
        if (l < L_LAYERS - 1) {
            k_zero<<<2, D_DIM, 0, stream>>>(sums, 2 * D_DIM);
            k_bn_reduce<<<cdiv(N, BN_ROWS), D_DIM, 0, stream>>>(aggF, sums, N);
            k_bn_apply<<<cdiv(N, BN_ROWS), D_DIM, 0, stream>>>(
                aggF, sums, gamma, beta, outBuf, N, l);
        }
    }
}
